// SelectiveSSM_12790412607789
// MI455X (gfx1250) — matrix operand order repaired and checked by reading, not compiled, not run on hardware
//
#include <hip/hip_runtime.h>

// ---------------------------------------------------------------------------
// Selective SSM (Mamba-style) for MI455X / gfx1250.
//   Phase 1: bf16 WMMA GEMM (4x2 register blocking) -> delta/softplus, Bm, Cm
//   Phase 2: 3-pass chunked parallel scan, v_pk_fma_f32 math,
//            async-to-LDS staging of the shared B/C chunk rows.
// ---------------------------------------------------------------------------

typedef __bf16 bf16_t;
typedef bf16_t v16bf __attribute__((ext_vector_type(16)));
typedef bf16_t v8bf  __attribute__((ext_vector_type(8)));
typedef float  v8f   __attribute__((ext_vector_type(8)));
typedef float  v2f   __attribute__((ext_vector_type(2)));

// explicit global-address-space pointers: force global_load/global_store
// (generic pointers through multi-source phis were lowering to flat_load_*,
//  which ties up DScnt and the LDS path on CDNA5)
#define AS1 __attribute__((address_space(1)))
typedef const AS1 unsigned short* gcu16;
typedef const AS1 v8bf*           gcv8bf;
typedef AS1 float*                gf32;

constexpr int BATCH  = 4;
constexpr int SEQ    = 4096;
constexpr int DMODEL = 1024;
constexpr int DSTATE = 16;
constexpr int BS     = BATCH * SEQ;        // 16384 GEMM rows
constexpr int NC     = 32;                 // scan chunks
constexpr int CL     = SEQ / NC;           // 128 steps per chunk
constexpr int MT     = 4;                  // row tiles per wave
constexpr int NT     = 2;                  // col tiles per wave
constexpr int RB     = BS / (16 * MT);     // 256 row blocks
constexpr int CB     = DMODEL / 16 / NT + 1; // 32 delta col blocks + 1 (Bm,Cm)
constexpr int NWAVE  = RB * CB;            // 8448 waves (divisible by 8)

__device__ __forceinline__ unsigned short f2bf(float f) {
  unsigned int u = __float_as_uint(f);
  u = (u + 0x7FFFu + ((u >> 16) & 1u)) >> 16;   // round-to-nearest-even bf16
  return (unsigned short)u;
}

// CDNA5 async global->LDS copy (ASYNCcnt path, CDNA5 ISA)
__device__ __forceinline__ void async_b128_to_lds(unsigned lds_off, const void* g) {
  asm volatile("global_load_async_to_lds_b128 %0, %1, off"
               :: "v"(lds_off), "v"(g) : "memory");
}
__device__ __forceinline__ void wait_async() {
  asm volatile("s_wait_asynccnt 0x0" ::: "memory");
}

// ---- convert f32 -> bf16 (row-major passthrough) --------------------------
__global__ void k_f32_to_bf16(const float* __restrict__ in,
                              unsigned short* __restrict__ out, int n) {
  int i = blockIdx.x * blockDim.x + threadIdx.x;
  if (i < n) out[i] = f2bf(in[i]);
}

// ---- transpose [K,N] f32 -> [N,K] bf16 ------------------------------------
__global__ void k_transpose_bf16(const float* __restrict__ W,
                                 unsigned short* __restrict__ Wt,
                                 int K, int N) {
  int i = blockIdx.x * blockDim.x + threadIdx.x;
  if (i < K * N) {
    int n = i / K;
    int k = i - n * K;
    Wt[i] = f2bf(W[k * N + n]);   // Wt[n][k] = W[k][n]
  }
}

// ---- fused projection GEMM: 4x2 16x16 tiles per wave ----------------------
// col blocks 0..31 -> delta = softplus(u*W_delta + b_delta)   [BS,1024]
// col block  32    -> (Bm, Cm) = u*W_B + b_B, u*W_C + b_C     [BS,16] each
__global__ __launch_bounds__(256)
void k_proj_wmma(const unsigned short* __restrict__ ubf,   // [BS,DMODEL] bf16
                 const unsigned short* __restrict__ wtd,   // [1024,1024] = W_delta^T
                 const unsigned short* __restrict__ wtb,   // [16,1024]   = W_B^T
                 const unsigned short* __restrict__ wtc,   // [16,1024]   = W_C^T
                 const float* __restrict__ bd,
                 const float* __restrict__ bb,
                 const float* __restrict__ bc,
                 float* __restrict__ delta,
                 float* __restrict__ Bm,
                 float* __restrict__ Cm) {
  const int wave = threadIdx.x >> 5;
  const int lane = threadIdx.x & 31;
  const int wid  = blockIdx.x * 8 + wave;
  const int cb   = wid % CB;
  const int rb   = wid / CB;
  const int g    = lane >> 4;    // half-wave group
  const int mr   = lane & 15;    // row (A frag) / col (B frag, C/D) in tile

  gcu16 Ap[MT];
  #pragma unroll
  for (int i = 0; i < MT; ++i)
    Ap[i] = (gcu16)(uintptr_t)(ubf + ((rb * MT + i) * 16 + mr) * DMODEL);

  gcu16 Bp[NT];
  if (cb < CB - 1) {
    Bp[0] = (gcu16)(uintptr_t)(wtd + ((cb * 2 + 0) * 16 + mr) * DMODEL);
    Bp[1] = (gcu16)(uintptr_t)(wtd + ((cb * 2 + 1) * 16 + mr) * DMODEL);
  } else {
    Bp[0] = (gcu16)(uintptr_t)(wtb + mr * DMODEL);
    Bp[1] = (gcu16)(uintptr_t)(wtc + mr * DMODEL);
  }

  v8f acc[MT][NT] = {};
  for (int k0 = 0; k0 < DMODEL; k0 += 32) {
    v16bf av[MT], bv[NT];
    #pragma unroll
    for (int i = 0; i < MT; ++i) {
      // A 16x32 bf16: lane m+16g holds K in [8g,8g+8) U [16+8g,16+8g+8)
      v8bf a0 = *(gcv8bf)(Ap[i] + k0 + 8 * g);
      v8bf a1 = *(gcv8bf)(Ap[i] + k0 + 16 + 8 * g);
      av[i] = __builtin_shufflevector(a0, a1, 0,1,2,3,4,5,6,7,8,9,10,11,12,13,14,15);
    }
    #pragma unroll
    for (int j = 0; j < NT; ++j) {
      // B 32x16 bf16: lane n+16g holds K in [8g,8g+8) U [16+8g,16+8g+8)
      v8bf b0 = *(gcv8bf)(Bp[j] + k0 + 8 * g);
      v8bf b1 = *(gcv8bf)(Bp[j] + k0 + 16 + 8 * g);
      bv[j] = __builtin_shufflevector(b0, b1, 0,1,2,3,4,5,6,7,8,9,10,11,12,13,14,15);
    }
    #pragma unroll
    for (int i = 0; i < MT; ++i)
      #pragma unroll
      for (int j = 0; j < NT; ++j)
        acc[i][j] = __builtin_amdgcn_wmma_f32_16x16x32_bf16(
            false, av[i], false, bv[j], (short)0, acc[i][j], false, false);
  }

  // C/D layout: lane n+16g, VGPR r -> row (r + 8g), col n
  gf32 dout = (gf32)(uintptr_t)delta;
  #pragma unroll
  for (int i = 0; i < MT; ++i) {
    const int row0 = (rb * MT + i) * 16 + 8 * g;
    #pragma unroll
    for (int j = 0; j < NT; ++j) {
      if (cb < CB - 1) {
        const int col = (cb * 2 + j) * 16 + mr;
        const float bias = bd[col];
        #pragma unroll
        for (int r = 0; r < 8; ++r) {
          float v = acc[i][j][r] + bias;
          v = (v > 20.0f) ? v : __logf(1.0f + __expf(v));   // softplus
          dout[(row0 + r) * DMODEL + col] = v;
        }
      } else {
        const float bias = (j == 0) ? bb[mr] : bc[mr];
        gf32 out = (gf32)(uintptr_t)((j == 0) ? Bm : Cm);
        #pragma unroll
        for (int r = 0; r < 8; ++r)
          out[(row0 + r) * DSTATE + mr] = acc[i][j][r] + bias;
      }
    }
  }
}

// ---- scan pass 1: per-chunk decay product + local end-state (h0 = 0) ------
__global__ __launch_bounds__(256)
void k_scan_local(const float* __restrict__ u,
                  const float* __restrict__ delta,
                  const float* __restrict__ Bm,
                  const float* __restrict__ A,
                  float* __restrict__ Aw,   // [NC, B*D, N] chunk decay product
                  float* __restrict__ Xw) { // [NC, B*D, N] chunk local h_end
  __shared__ __align__(16) float sB[CL * DSTATE];   // 8 KB, shared by block

  const int tid  = threadIdx.x;
  const int bgid = blockIdx.x * 256;
  const int c    = bgid >> 12;               // chunk (uniform per block)
  const int b    = (bgid & 4095) >> 10;      // batch (uniform per block)
  const int d    = ((bgid & 4095) & 1023) + tid;
  const int gid  = bgid + tid;

  const int base_b = (b * SEQ + c * CL) * DSTATE;   // chunk B-row base

  // stage the chunk's Bm rows into LDS via CDNA5 async copy (2048 floats)
  {
    const unsigned sb = (unsigned)(uintptr_t)sB + tid * 32u;
    const float* src = Bm + base_b + tid * 8;
    async_b128_to_lds(sb,      src);
    async_b128_to_lds(sb + 16, src + 4);
    wait_async();
    __syncthreads();
  }

  const v2f* Arow = (const v2f*)(A + d * DSTATE);
  v2f a_d[8], h[8], ap[8];
  #pragma unroll
  for (int n = 0; n < 8; ++n) {
    a_d[n] = Arow[n];
    h[n]   = (v2f){0.0f, 0.0f};
    ap[n]  = (v2f){1.0f, 1.0f};
  }

  const int base_u = (b * SEQ + c * CL) * DMODEL + d;
  const v2f one2 = {1.0f, 1.0f};

  for (int t = 0; t < CL; ++t) {
    const float uu = u[base_u + t * DMODEL];
    const float dl = delta[base_u + t * DMODEL];
    const float du = dl * uu;
    const v2f dl2 = {dl, dl}, du2 = {du, du};
    const v2f* Brow = (const v2f*)(sB + t * DSTATE);
    #pragma unroll
    for (int n = 0; n < 8; ++n) {
      const v2f a = __builtin_elementwise_fma(dl2, a_d[n], one2);  // v_pk_fma_f32
      ap[n] *= a;
      h[n] = __builtin_elementwise_fma(a, h[n], du2 * Brow[n]);
    }
  }

  v2f* awp = (v2f*)(Aw + gid * DSTATE);
  v2f* xwp = (v2f*)(Xw + gid * DSTATE);
  #pragma unroll
  for (int n = 0; n < 8; ++n) { awp[n] = ap[n]; xwp[n] = h[n]; }
}

// ---- scan pass 2: stitch carries across chunks ----------------------------
__global__ __launch_bounds__(256)
void k_scan_stitch(const float* __restrict__ Aw,
                   const float* __restrict__ Xw,
                   float* __restrict__ Hw) {  // [NC, B*D, N] chunk start state
  const int j = blockIdx.x * blockDim.x + threadIdx.x;  // B*D*N/2 = 32768
  const int stride = BATCH * DMODEL * DSTATE / 2;
  const v2f* aw = (const v2f*)Aw;
  const v2f* xw = (const v2f*)Xw;
  v2f* hw = (v2f*)Hw;
  v2f hs = {0.0f, 0.0f};
  for (int c = 0; c < NC; ++c) {
    const int idx = c * stride + j;
    hw[idx] = hs;
    hs = __builtin_elementwise_fma(aw[idx], hs, xw[idx]);
  }
}

// ---- scan pass 3: re-scan chunks from true start state, emit y = C.h ------
__global__ __launch_bounds__(256)
void k_scan_final(const float* __restrict__ u,
                  const float* __restrict__ delta,
                  const float* __restrict__ Bm,
                  const float* __restrict__ Cm,
                  const float* __restrict__ A,
                  const float* __restrict__ Hw,
                  float* __restrict__ y) {
  __shared__ __align__(16) float sB[CL * DSTATE];
  __shared__ __align__(16) float sC[CL * DSTATE];

  const int tid  = threadIdx.x;
  const int bgid = blockIdx.x * 256;
  const int c    = bgid >> 12;
  const int b    = (bgid & 4095) >> 10;
  const int d    = ((bgid & 4095) & 1023) + tid;
  const int gid  = bgid + tid;

  const int base_b = (b * SEQ + c * CL) * DSTATE;

  {
    const unsigned sb = (unsigned)(uintptr_t)sB + tid * 32u;
    const unsigned sc = (unsigned)(uintptr_t)sC + tid * 32u;
    const float* srcB = Bm + base_b + tid * 8;
    const float* srcC = Cm + base_b + tid * 8;
    async_b128_to_lds(sb,      srcB);
    async_b128_to_lds(sb + 16, srcB + 4);
    async_b128_to_lds(sc,      srcC);
    async_b128_to_lds(sc + 16, srcC + 4);
    wait_async();
    __syncthreads();
  }

  const v2f* Arow = (const v2f*)(A + d * DSTATE);
  const v2f* hwp  = (const v2f*)(Hw + gid * DSTATE);
  v2f a_d[8], h[8];
  #pragma unroll
  for (int n = 0; n < 8; ++n) { a_d[n] = Arow[n]; h[n] = hwp[n]; }

  const int base_u = (b * SEQ + c * CL) * DMODEL + d;
  const v2f one2 = {1.0f, 1.0f};

  for (int t = 0; t < CL; ++t) {
    const float uu = u[base_u + t * DMODEL];
    const float dl = delta[base_u + t * DMODEL];
    const float du = dl * uu;
    const v2f dl2 = {dl, dl}, du2 = {du, du};
    const v2f* Brow = (const v2f*)(sB + t * DSTATE);
    const v2f* Crow = (const v2f*)(sC + t * DSTATE);
    v2f acc2 = {0.0f, 0.0f};
    #pragma unroll
    for (int n = 0; n < 8; ++n) {
      const v2f a = __builtin_elementwise_fma(dl2, a_d[n], one2);
      h[n] = __builtin_elementwise_fma(a, h[n], du2 * Brow[n]);
      acc2 = __builtin_elementwise_fma(Crow[n], h[n], acc2);
    }
    y[base_u + t * DMODEL] = acc2[0] + acc2[1];
  }
}

// ---------------------------------------------------------------------------
extern "C" void kernel_launch(void* const* d_in, const int* in_sizes, int n_in,
                              void* d_out, int out_size, void* d_ws, size_t ws_size,
                              hipStream_t stream) {
  const float* u  = (const float*)d_in[0];
  const float* Wd = (const float*)d_in[1];
  const float* bd = (const float*)d_in[2];
  const float* Wb = (const float*)d_in[3];
  const float* bb = (const float*)d_in[4];
  const float* Wc = (const float*)d_in[5];
  const float* bc = (const float*)d_in[6];
  const float* A  = (const float*)d_in[7];
  float* y = (float*)d_out;

  // workspace layout (bytes)
  char* ws = (char*)d_ws;
  size_t off = 0;
  unsigned short* ubf = (unsigned short*)(ws + off); off += (size_t)BS * DMODEL * 2;      // 32 MB
  unsigned short* wtd = (unsigned short*)(ws + off); off += (size_t)DMODEL * DMODEL * 2;  //  2 MB
  unsigned short* wtb = (unsigned short*)(ws + off); off += (size_t)DSTATE * DMODEL * 2;
  unsigned short* wtc = (unsigned short*)(ws + off); off += (size_t)DSTATE * DMODEL * 2;
  float* delta = (float*)(ws + off); off += (size_t)BS * DMODEL * 4;                      // 64 MB
  float* Bm    = (float*)(ws + off); off += (size_t)BS * DSTATE * 4;                      //  1 MB
  float* Cm    = (float*)(ws + off); off += (size_t)BS * DSTATE * 4;                      //  1 MB
  float* Aw    = (float*)(ws + off); off += (size_t)NC * BATCH * DMODEL * DSTATE * 4;     //  8 MB
  float* Xw    = (float*)(ws + off); off += (size_t)NC * BATCH * DMODEL * DSTATE * 4;     //  8 MB
  float* Hw    = (float*)(ws + off); off += (size_t)NC * BATCH * DMODEL * DSTATE * 4;     //  8 MB

  const int nu = BS * DMODEL;

  // 0) precision conversion + weight transposes (bf16 operand staging)
  k_f32_to_bf16<<<(nu + 255) / 256, 256, 0, stream>>>(u, ubf, nu);
  k_transpose_bf16<<<(DMODEL * DMODEL + 255) / 256, 256, 0, stream>>>(Wd, wtd, DMODEL, DMODEL);
  k_transpose_bf16<<<(DMODEL * DSTATE + 255) / 256, 256, 0, stream>>>(Wb, wtb, DMODEL, DSTATE);
  k_transpose_bf16<<<(DMODEL * DSTATE + 255) / 256, 256, 0, stream>>>(Wc, wtc, DMODEL, DSTATE);

  // 1) fused WMMA projection GEMM (delta/softplus + Bm + Cm), 4x2 blocking
  k_proj_wmma<<<NWAVE / 8, 256, 0, stream>>>(ubf, wtd, wtb, wtc, bd, bb, bc,
                                             delta, Bm, Cm);

  // 2) chunked parallel scan
  const int nscan = NC * BATCH * DMODEL;          // 131072 threads
  k_scan_local <<<nscan / 256, 256, 0, stream>>>(u, delta, Bm, A, Aw, Xw);
  k_scan_stitch<<<(BATCH * DMODEL * DSTATE / 2) / 256, 256, 0, stream>>>(Aw, Xw, Hw);
  k_scan_final <<<nscan / 256, 256, 0, stream>>>(u, delta, Bm, Cm, A, Hw, y);
}
